// AndAttention_52132313039254
// MI455X (gfx1250) — matrix operand order repaired and checked by reading, not compiled, not run on hardware
//
#include <hip/hip_runtime.h>
#include <hip/hip_bf16.h>
#include <stdint.h>

// ---------------------------------------------------------------------------
// Problem constants (reference: B=16384, D=1024, 4 hidden layers of 2048)
// ---------------------------------------------------------------------------
#define BATCH 16384
#define DDIM  1024
#define D2    2048
#define NLAYERS 4

// GEMM tiling: 128(M) x 256(N) block tile, K-step 32, 8 waves (wave32).
// Waves arranged 2(M) x 4(N); each wave owns a 64x64 sub-tile = 4x4 WMMA
// accumulators, so every fragment feeds 4 back-to-back WMMAs.
#define TM 128
#define TN 256
#define TK 32
#define LDST 40   // padded LDS row stride in ushorts (80B, 16B-aligned, conflict-free)

typedef __attribute__((ext_vector_type(16))) __bf16 v16bf;
typedef __attribute__((ext_vector_type(8)))  float  v8f;

__device__ __forceinline__ unsigned short f2bf(float f) {
  unsigned int u = __builtin_bit_cast(unsigned int, f);
  u += 0x7FFFu + ((u >> 16) & 1u);   // round-to-nearest-even
  return (unsigned short)(u >> 16);
}

// Async DMA: global -> LDS, 16 bytes per lane, tracked by ASYNCcnt.
// INST_OFFSET is added to BOTH the LDS and the global address (ISA §10.7),
// so a 32B segment is two instructions sharing one address pair.
#define ASYNC_B128(lds32, gaddr64)                                          \
  asm volatile("global_load_async_to_lds_b128 %0, %1, off"                  \
               :: "v"(lds32), "v"(gaddr64) : "memory")
#define ASYNC_B128_OFF16(lds32, gaddr64)                                    \
  asm volatile("global_load_async_to_lds_b128 %0, %1, off offset:16"        \
               :: "v"(lds32), "v"(gaddr64) : "memory")
#define WAIT_ASYNC0() asm volatile("s_wait_asynccnt 0x0" ::: "memory")

// ---------------------------------------------------------------------------
// Weight conversion fp32 -> bf16 (once per launch; weights then serve all
// M-blocks of each layer out of the 192MB L2).
// ---------------------------------------------------------------------------
__global__ __launch_bounds__(256)
void f32_to_bf16_kernel(const float* __restrict__ src,
                        unsigned short* __restrict__ dst) {
  size_t i = ((size_t)blockIdx.x * 256 + threadIdx.x) * 8;
  float4 v0 = *(const float4*)(src + i);
  float4 v1 = *(const float4*)(src + i + 4);
  uint4 o;
  o.x = (unsigned)f2bf(v0.x) | ((unsigned)f2bf(v0.y) << 16);
  o.y = (unsigned)f2bf(v0.z) | ((unsigned)f2bf(v0.w) << 16);
  o.z = (unsigned)f2bf(v1.x) | ((unsigned)f2bf(v1.y) << 16);
  o.w = (unsigned)f2bf(v1.z) | ((unsigned)f2bf(v1.w) << 16);
  *(uint4*)(dst + i) = o;
}

// ---------------------------------------------------------------------------
// Fused 2-token attention + concat. One wave32 per batch row; registers hold
// the row, __shfl_xor reduces the three dots, stable 2x2 softmax, bf16 out.
// ---------------------------------------------------------------------------
__global__ __launch_bounds__(256)
void attn_fuse_kernel(const float* __restrict__ x1,
                      const float* __restrict__ x2,
                      unsigned short* __restrict__ h) {
  const int lane = threadIdx.x & 31;
  const int wave = threadIdx.x >> 5;
  const int row  = blockIdx.x * 8 + wave;

  const float4* p1 = (const float4*)(x1 + (size_t)row * DDIM);
  const float4* p2 = (const float4*)(x2 + (size_t)row * DDIM);

  float4 a[8], b[8];
  float s11 = 0.f, s12 = 0.f, s22 = 0.f;
#pragma unroll
  for (int i = 0; i < 8; ++i) {
    a[i] = p1[lane + 32 * i];
    b[i] = p2[lane + 32 * i];
    s11 += a[i].x * a[i].x + a[i].y * a[i].y + a[i].z * a[i].z + a[i].w * a[i].w;
    s12 += a[i].x * b[i].x + a[i].y * b[i].y + a[i].z * b[i].z + a[i].w * b[i].w;
    s22 += b[i].x * b[i].x + b[i].y * b[i].y + b[i].z * b[i].z + b[i].w * b[i].w;
  }
#pragma unroll
  for (int off = 16; off > 0; off >>= 1) {
    s11 += __shfl_xor(s11, off, 32);
    s12 += __shfl_xor(s12, off, 32);
    s22 += __shfl_xor(s22, off, 32);
  }
  const float inv_t = 1.0f / 32.0f;   // TEMPERATURE = 32
  const float l11 = s11 * inv_t, l12 = s12 * inv_t, l22 = s22 * inv_t;
  float m0 = fmaxf(l11, l12);
  float e00 = __expf(l11 - m0), e01 = __expf(l12 - m0);
  float r0 = 1.0f / (e00 + e01);
  float a00 = e00 * r0, a01 = e01 * r0;
  float m1 = fmaxf(l12, l22);
  float e10 = __expf(l12 - m1), e11 = __expf(l22 - m1);
  float r1 = 1.0f / (e10 + e11);
  float a10 = e10 * r1, a11 = e11 * r1;

  unsigned short* h0 = h + (size_t)row * D2;       // attended token 0
  unsigned short* h1 = h0 + DDIM;                  // attended token 1
#pragma unroll
  for (int i = 0; i < 8; ++i) {
    const int idx = (lane + 32 * i) * 4;
    uint2 pk0, pk1;
    pk0.x = (unsigned)f2bf(a00 * a[i].x + a01 * b[i].x) |
            ((unsigned)f2bf(a00 * a[i].y + a01 * b[i].y) << 16);
    pk0.y = (unsigned)f2bf(a00 * a[i].z + a01 * b[i].z) |
            ((unsigned)f2bf(a00 * a[i].w + a01 * b[i].w) << 16);
    pk1.x = (unsigned)f2bf(a10 * a[i].x + a11 * b[i].x) |
            ((unsigned)f2bf(a10 * a[i].y + a11 * b[i].y) << 16);
    pk1.y = (unsigned)f2bf(a10 * a[i].z + a11 * b[i].z) |
            ((unsigned)f2bf(a10 * a[i].w + a11 * b[i].w) << 16);
    *(uint2*)(h0 + idx) = pk0;
    *(uint2*)(h1 + idx) = pk1;
  }
}

// ---------------------------------------------------------------------------
// Fragment gathers from LDS following CDNA5 16-bit WMMA VGPR layouts.
// A 16x32 (MxK): lane L<16 -> M=L, K = 0..7 & 16..23 ; lane L>=16 -> K = 8..15 & 24..31
// B 32x16 (KxN): lane L<16 -> N=L%16, K = 0..7 & 16..23 ; lane L>=16 -> K = 8..15 & 24..31
// ---------------------------------------------------------------------------
union FragU { v16bf v; uint4 q[2]; };

__device__ __forceinline__ v16bf ld_afrag(const unsigned short* sA, int mBase, int lane) {
  const int r = mBase + (lane & 15);
  const int k = (lane >> 4) << 3;          // 0 or 8
  const unsigned short* p = sA + r * LDST + k;
  FragU f;
  f.q[0] = *(const uint4*)p;               // K = k .. k+7
  f.q[1] = *(const uint4*)(p + 16);        // K = k+16 .. k+23
  return f.v;
}

__device__ __forceinline__ v16bf ld_bfrag(const unsigned short* sB, int nBase, int lane) {
  const int r = nBase + (lane & 15);
  const int k = (lane >> 4) << 3;          // 0 or 8
  const unsigned short* p = sB + r * LDST + k;
  FragU f;
  f.q[0] = *(const uint4*)p;               // K = k .. k+7
  f.q[1] = *(const uint4*)(p + 16);        // K = k+16 .. k+23
  return f.v;
}

// ---------------------------------------------------------------------------
// out[M,N] = act(A[M,K] @ W[N,K]^T + bias), A/W in bf16, fp32 accumulate.
// Staging uses GLOBAL_LOAD_ASYNC_TO_LDS_B128 (ASYNCcnt) -- no VGPR roundtrip;
// DMA of buffer (cur^1) overlaps the WMMA chain on buffer (cur).
// ---------------------------------------------------------------------------
template <bool RELU, bool OUT_BF16>
__global__ __launch_bounds__(256)
void gemm_bias_act_kernel(const unsigned short* __restrict__ A,
                          const unsigned short* __restrict__ W,
                          const float* __restrict__ bias,
                          void* __restrict__ out,
                          int M, int N, int K) {
  __shared__ unsigned short sA[2][TM * LDST];   // 2 x 10240B
  __shared__ unsigned short sB[2][TN * LDST];   // 2 x 20480B   (60KB total)

  const int tid   = threadIdx.x;
  const int lane  = tid & 31;
  const int wave  = tid >> 5;
  const int waveM = wave & 1;          // 0..1 -> 64 rows each
  const int waveN = wave >> 1;         // 0..3 -> 64 cols each
  const int mBlk  = blockIdx.y * TM;
  const int nBlk  = blockIdx.x * TN;

  // Staging map: thread t moves 16 contiguous bf16 (32B) of A-row (t/2) and
  // of W-rows (t/2) and (t/2 + 128) per K-tile, via 2 async B128 ops each.
  const int srow = tid >> 1;           // 0..127
  const int sseg = (tid & 1) << 4;     // 0 or 16
  const unsigned short* gA  = A + (size_t)(mBlk + srow) * K + sseg;
  const unsigned short* gW0 = W + (size_t)(nBlk + srow) * K + sseg;
  const unsigned short* gW1 = W + (size_t)(nBlk + srow + 128) * K + sseg;

  // LDS byte offsets (low 32 bits of the flat address ARE the LDS offset:
  // the shared aperture lives entirely in addr[63:32]).
  const unsigned laBase  = (unsigned)(uintptr_t)&sA[0][srow * LDST + sseg];
  const unsigned lb0Base = (unsigned)(uintptr_t)&sB[0][srow * LDST + sseg];
  const unsigned lb1Base = (unsigned)(uintptr_t)&sB[0][(srow + 128) * LDST + sseg];
  const unsigned aBufStride = (unsigned)(TM * LDST * sizeof(unsigned short));
  const unsigned bBufStride = (unsigned)(TN * LDST * sizeof(unsigned short));

  const int nk = K / TK;

  auto stage = [&](int buf, int kt) {
    const unsigned la  = laBase  + (unsigned)buf * aBufStride;
    const unsigned lb0 = lb0Base + (unsigned)buf * bBufStride;
    const unsigned lb1 = lb1Base + (unsigned)buf * bBufStride;
    const unsigned long long ga  = (unsigned long long)(uintptr_t)(gA  + (size_t)kt * TK);
    const unsigned long long gb0 = (unsigned long long)(uintptr_t)(gW0 + (size_t)kt * TK);
    const unsigned long long gb1 = (unsigned long long)(uintptr_t)(gW1 + (size_t)kt * TK);
    // Speculative prefetch of the K+2 tiles toward L2 (global_prefetch_b8).
    __builtin_prefetch(gA  + (size_t)(kt + 2) * TK, 0, 1);
    __builtin_prefetch(gW0 + (size_t)(kt + 2) * TK, 0, 1);
    __builtin_prefetch(gW1 + (size_t)(kt + 2) * TK, 0, 1);
    ASYNC_B128(la, ga);        ASYNC_B128_OFF16(la, ga);
    ASYNC_B128(lb0, gb0);      ASYNC_B128_OFF16(lb0, gb0);
    ASYNC_B128(lb1, gb1);      ASYNC_B128_OFF16(lb1, gb1);
  };

  v8f acc[4][4];
  const v8f vzero = {0.f, 0.f, 0.f, 0.f, 0.f, 0.f, 0.f, 0.f};
#pragma unroll
  for (int mi = 0; mi < 4; ++mi)
#pragma unroll
    for (int ni = 0; ni < 4; ++ni) acc[mi][ni] = vzero;

  stage(0, 0);
  int cur = 0;
  for (int kt = 0; kt < nk; ++kt) {
    WAIT_ASYNC0();        // this wave's async writes into buffer `cur` landed
    __syncthreads();      // ... and everyone else's too -> publish buffer
    if (kt + 1 < nk) stage(cur ^ 1, kt + 1);  // DMA next tile during compute

    v16bf af[4], bfv[4];
#pragma unroll
    for (int mi = 0; mi < 4; ++mi)
      af[mi] = ld_afrag(sA[cur], waveM * 64 + mi * 16, lane);
#pragma unroll
    for (int ni = 0; ni < 4; ++ni)
      bfv[ni] = ld_bfrag(sB[cur], waveN * 64 + ni * 16, lane);

#pragma unroll
    for (int mi = 0; mi < 4; ++mi)
#pragma unroll
      for (int ni = 0; ni < 4; ++ni)
        acc[mi][ni] = __builtin_amdgcn_wmma_f32_16x16x32_bf16(
            false, af[mi], false, bfv[ni], (short)0, acc[mi][ni], false, false);

    cur ^= 1;
  }

  // Epilogue: C/D layout -> lane l, reg r holds (M = r + 8*(l/16), N = l%16).
#pragma unroll
  for (int ni = 0; ni < 4; ++ni) {
    const int col = nBlk + waveN * 64 + ni * 16 + (lane & 15);
    const float bv = bias[col];
#pragma unroll
    for (int mi = 0; mi < 4; ++mi) {
      const int row0 = mBlk + waveM * 64 + mi * 16 + ((lane >> 4) << 3);
#pragma unroll
      for (int r = 0; r < 8; ++r) {
        float v = acc[mi][ni][r] + bv;
        if (RELU) v = fmaxf(v, 0.0f);
        const size_t o = (size_t)(row0 + r) * N + col;
        if (OUT_BF16) ((unsigned short*)out)[o] = f2bf(v);
        else          ((float*)out)[o] = v;
      }
    }
  }
}

// ---------------------------------------------------------------------------
// Launch: attention -> 4x (GEMM+ReLU, bf16 ping-pong) -> final GEMM (fp32 out)
// Workspace layout:
//   hA   : 16384*2048*2  = 64 MB
//   hB   : 16384*2048*2  = 64 MB
//   Wbf  : 4*2048*2048*2 = 32 MB
//   Wlbf : 1024*2048*2   =  4 MB     (total ~164 MB)
// ---------------------------------------------------------------------------
extern "C" void kernel_launch(void* const* d_in, const int* in_sizes, int n_in,
                              void* d_out, int out_size, void* d_ws, size_t ws_size,
                              hipStream_t stream) {
  (void)in_sizes; (void)n_in; (void)out_size; (void)ws_size;

  const float* x1 = (const float*)d_in[0];
  const float* x2 = (const float*)d_in[1];
  const float* Ws = (const float*)d_in[2];   // [4, 2048, 2048]
  const float* bs = (const float*)d_in[3];   // [4, 2048]
  const float* Wl = (const float*)d_in[4];   // [1024, 2048]
  const float* bl = (const float*)d_in[5];   // [1024]
  float* out = (float*)d_out;                // [16384, 1024]

  unsigned short* hA   = (unsigned short*)d_ws;
  unsigned short* hB   = hA + (size_t)BATCH * D2;
  unsigned short* Wbf  = hB + (size_t)BATCH * D2;
  unsigned short* Wlbf = Wbf + (size_t)NLAYERS * D2 * D2;

  // Weight conversion fp32 -> bf16
  {
    const size_t n1 = (size_t)NLAYERS * D2 * D2;     // 16,777,216
    f32_to_bf16_kernel<<<dim3((unsigned)(n1 / (8 * 256))), dim3(256), 0, stream>>>(Ws, Wbf);
    const size_t n2 = (size_t)DDIM * D2;             // 2,097,152
    f32_to_bf16_kernel<<<dim3((unsigned)(n2 / (8 * 256))), dim3(256), 0, stream>>>(Wl, Wlbf);
  }

  // Attention + concat -> hA (bf16)
  attn_fuse_kernel<<<dim3(BATCH / 8), dim3(256), 0, stream>>>(x1, x2, hA);

  const dim3 blk(256);
  const dim3 gHidden(D2 / TN, BATCH / TM);   // 8 x 128 blocks
  gemm_bias_act_kernel<true, true ><<<gHidden, blk, 0, stream>>>(hA, Wbf + 0ull * D2 * D2, bs + 0 * D2, hB, BATCH, D2, D2);
  gemm_bias_act_kernel<true, true ><<<gHidden, blk, 0, stream>>>(hB, Wbf + 1ull * D2 * D2, bs + 1 * D2, hA, BATCH, D2, D2);
  gemm_bias_act_kernel<true, true ><<<gHidden, blk, 0, stream>>>(hA, Wbf + 2ull * D2 * D2, bs + 2 * D2, hB, BATCH, D2, D2);
  gemm_bias_act_kernel<true, true ><<<gHidden, blk, 0, stream>>>(hB, Wbf + 3ull * D2 * D2, bs + 3 * D2, hA, BATCH, D2, D2);

  const dim3 gLast(DDIM / TN, BATCH / TM);   // 4 x 128 blocks
  gemm_bias_act_kernel<false, false><<<gLast, blk, 0, stream>>>(hA, Wlbf, bl, out, BATCH, DDIM, D2);
}
